// GPTXAttention_65695819760161
// MI455X (gfx1250) — matrix operand order repaired and checked by reading, not compiled, not run on hardware
//
#include <hip/hip_runtime.h>
#include <hip/hip_bf16.h>

// ---------------- problem constants ----------------
constexpr int Bc  = 2;
constexpr int Tc  = 4096;
constexpr int Dc  = 2048;
constexpr int Hc  = 16;
constexpr int HDc = 128;            // Dc / Hc
constexpr int NTc = Bc * Tc;        // 8192 tokens
constexpr int NC    = 32;           // scan chunks per (b,h) chain
constexpr int CHUNK = Tc / NC;      // 128
// OFFSET = -log(sqrt(2048)) + 1e-6
constexpr float OFFSETF = -3.8123084930796992f;

typedef __bf16 bf16_t;
typedef __attribute__((ext_vector_type(16))) __bf16 v16bf;
typedef __attribute__((ext_vector_type(8)))  float  v8f;

// pointer types for the async-DMA builtin (param type per hipcc diagnostic:
// "__vector_size__(4*sizeof(int)) int __device__ *" == v4i addrspace(1)*)
typedef int v4i_t __attribute__((vector_size(16)));
typedef __attribute__((address_space(1))) v4i_t* gv4i_p;
typedef __attribute__((address_space(3))) v4i_t* lv4i_p;

// ---------------- CDNA5 async global->LDS path ----------------
#if defined(__has_builtin)
#if __has_builtin(__builtin_amdgcn_global_load_async_to_lds_b128)
#define HAVE_ASYNC_LDS 1
#endif
#if __has_builtin(__builtin_amdgcn_s_wait_asynccnt)
#define HAVE_WAIT_ASYNC 1
#endif
#endif

__device__ __forceinline__ void async_wait0() {
#if defined(HAVE_WAIT_ASYNC)
  __builtin_amdgcn_s_wait_asynccnt(0);
#else
  asm volatile("s_wait_asynccnt 0" ::: "memory");
#endif
}

// copy 32 bytes global -> LDS (per thread)
__device__ __forceinline__ void stage32B(const bf16_t* __restrict__ g, bf16_t* l) {
#if defined(HAVE_ASYNC_LDS)
  // INST_OFFSET applies to BOTH global and LDS address (ISA 15.18.3 op 98)
  gv4i_p gp = (gv4i_p)(void*)const_cast<bf16_t*>(g);
  lv4i_p lp = (lv4i_p)(void*)l;
  __builtin_amdgcn_global_load_async_to_lds_b128(gp, lp, 0, 0);
  __builtin_amdgcn_global_load_async_to_lds_b128(gp, lp, 16, 0);
#else
  uint4 a0 = *(const uint4*)g;
  uint4 a1 = *(const uint4*)(g + 8);
  *(uint4*)l       = a0;
  *(uint4*)(l + 8) = a1;
#endif
}

// ---------------- f32 -> bf16 convert (vectorized x4) ----------------
__global__ __launch_bounds__(256) void cvt_bf16_kernel(const float* __restrict__ in,
                                                       bf16_t* __restrict__ out, int n4) {
  int i = blockIdx.x * blockDim.x + threadIdx.x;
  if (i < n4) {
    float4 v = ((const float4*)in)[i];
    alignas(8) bf16_t o[4] = {(bf16_t)v.x, (bf16_t)v.y, (bf16_t)v.z, (bf16_t)v.w};
    *(uint2*)(out + (size_t)i * 4) = *(const uint2*)o;
  }
}

// ---------------- f32 [K][N] -> bf16 [N][K] transpose-convert ----------------
__global__ __launch_bounds__(256) void cvtT_bf16_kernel(const float* __restrict__ W,
                                                        bf16_t* __restrict__ Wt, int Dd) {
  __shared__ float tile[32][33];
  const int tx = threadIdx.x & 31;
  const int ty = threadIdx.x >> 5;        // 0..7
  const int nb = blockIdx.x * 32;         // n tile
  const int kb = blockIdx.y * 32;         // k tile
#pragma unroll
  for (int r = 0; r < 4; ++r)
    tile[ty + r * 8][tx] = W[(size_t)(kb + ty + r * 8) * Dd + nb + tx];
  __syncthreads();
#pragma unroll
  for (int r = 0; r < 4; ++r)
    Wt[(size_t)(nb + ty + r * 8) * Dd + kb + tx] = (bf16_t)tile[tx][ty + r * 8];
}

// ---------------- bf16 WMMA GEMM: C[MxN] = A[MxK] * Bt[NxK]^T ----------------
// MODE 0: f32 out, MODE 1: bf16 out, MODE 2: f32 exp(x + offs) out
template <int MODE>
__global__ __launch_bounds__(256) void gemm_bf16_wmma(const bf16_t* __restrict__ A,
                                                      const bf16_t* __restrict__ Bt,
                                                      void* __restrict__ Cout,
                                                      int M, int N, int K, float offs) {
  constexpr int TM = 128, TN = 128, TK = 32;
  __shared__ bf16_t As[2][TM * TK];       // 2 x 8 KB
  __shared__ bf16_t Bs[2][TN * TK];       // 2 x 8 KB

  const int tid   = threadIdx.x;
  const int lane  = tid & 31;
  const int wave  = tid >> 5;             // 0..7
  const int waveM = wave >> 2;            // 0..1 -> 64 rows each
  const int waveN = wave & 3;             // 0..3 -> 32 cols each
  const int half  = lane >> 4;            // 0/1
  const int rlo   = lane & 15;

  const int m0 = blockIdx.y * TM;
  const int n0 = blockIdx.x * TN;

  // cooperative staging: thread i -> row i/2, 16 bf16 (32B) half (i&1)
  const int lrow = tid >> 1;
  const int lcol = (tid & 1) * 16;

  v8f acc[4][2] = {};                     // 8 x 16x16 f32 accumulators

  const bf16_t* gA = A  + (size_t)(m0 + lrow) * K + lcol;
  const bf16_t* gB = Bt + (size_t)(n0 + lrow) * K + lcol;
  bf16_t* lA = &As[0][lrow * TK + lcol];  // buffer 0 base; buffer 1 = +TM*TK
  bf16_t* lB = &Bs[0][lrow * TK + lcol];

  // prologue: fill buffer 0 with K-slab 0
  stage32B(gA, lA);
  stage32B(gB, lB);
  async_wait0();
  __syncthreads();

  int cur = 0;
  for (int k0 = 0; k0 < K; k0 += TK) {
    // kick off async fill of the next K-slab into the other buffer
    if (k0 + TK < K) {
      const int nxt = cur ^ 1;
      stage32B(gA + k0 + TK, lA + nxt * (TM * TK));
      stage32B(gB + k0 + TK, lB + nxt * (TN * TK));
      if (k0 + 2 * TK < K) {              // warm L2 for the slab after that
        __builtin_prefetch(gA + k0 + 2 * TK, 0, 1);
        __builtin_prefetch(gB + k0 + 2 * TK, 0, 1);
      }
    }

    // ---- compute on current buffer ----
    union Frag { uint4 u[2]; v16bf v; };
    Frag bf[2];
#pragma unroll
    for (int nt = 0; nt < 2; ++nt) {
      const bf16_t* p = &Bs[cur][(waveN * 32 + nt * 16 + rlo) * TK + half * 16];
      bf[nt].u[0] = *(const uint4*)p;
      bf[nt].u[1] = *(const uint4*)(p + 8);
    }
#pragma unroll
    for (int mt = 0; mt < 4; ++mt) {
      const bf16_t* p = &As[cur][(waveM * 64 + mt * 16 + rlo) * TK];
      Frag af;
      af.u[0] = *(const uint4*)(p + half * 16);
      af.u[1] = *(const uint4*)(p + half * 16 + 8);
#pragma unroll
      for (int nt = 0; nt < 2; ++nt) {
        acc[mt][nt] = __builtin_amdgcn_wmma_f32_16x16x32_bf16(
            false, af.v, false, bf[nt].v, (short)0, acc[mt][nt], false, false);
      }
    }

    async_wait0();       // my async fills of next buffer are in LDS
    __syncthreads();     // everyone done reading cur + filling nxt
    cur ^= 1;
  }

  // epilogue: VGPR j -> row = base + (lane>>4)*8 + j, col = base + (lane&15)
#pragma unroll
  for (int mt = 0; mt < 4; ++mt) {
#pragma unroll
    for (int nt = 0; nt < 2; ++nt) {
      const int col     = n0 + waveN * 32 + nt * 16 + rlo;
      const int rowbase = m0 + waveM * 64 + mt * 16 + half * 8;
#pragma unroll
      for (int j = 0; j < 8; ++j) {
        float v = acc[mt][nt][j];
        size_t idx = (size_t)(rowbase + j) * N + col;
        if constexpr (MODE == 0)      ((float*)Cout)[idx]  = v;
        else if constexpr (MODE == 1) ((bf16_t*)Cout)[idx] = (bf16_t)v;
        else                          ((float*)Cout)[idx]  = __expf(v + offs);
      }
    }
  }
}

// ---------------- dot[b,h,t] = sum_d Qp*Kp (one wave32 per dot) ----------------
__global__ __launch_bounds__(256) void dot_kernel(const float* __restrict__ Qp,
                                                  const float* __restrict__ Kp,
                                                  float* __restrict__ dotv) {
  const int w    = blockIdx.x * 8 + (threadIdx.x >> 5); // wave id = (b*H+h)*T + t
  const int lane = threadIdx.x & 31;
  const int t  = w % Tc;
  const int bh = w / Tc;
  const int b  = bh / Hc;
  const int h  = bh % Hc;
  const size_t base = (size_t)(b * Tc + t) * Dc + h * HDc + lane * 4;
  const float4 q = *(const float4*)(Qp + base);
  const float4 k = *(const float4*)(Kp + base);
  float s = q.x * k.x + q.y * k.y + q.z * k.z + q.w * k.w;
#pragma unroll
  for (int o = 16; o >= 1; o >>= 1) s += __shfl_xor(s, o, 32);
  if (lane == 0) dotv[w] = s;
}

// ---------------- blocked scan pass 1: chunk partial sums ----------------
__global__ __launch_bounds__(128) void scan_partial_kernel(const float* __restrict__ dotv,
                                                           const float* __restrict__ Vf,
                                                           float* __restrict__ psum) {
  const int blk = blockIdx.x;
  const int c  = blk % NC;
  const int bh = blk / NC;
  const int b  = bh / Hc;
  const int h  = bh % Hc;
  const int d  = threadIdx.x;
  float s = 0.f;
  const int tbase = c * CHUNK;
#pragma unroll 4
  for (int tt = 0; tt < CHUNK; ++tt) {
    const int t = tbase + tt;
    s += dotv[(size_t)bh * Tc + t] * Vf[(size_t)(b * Tc + t) * Dc + h * HDc + d];
  }
  psum[((size_t)bh * NC + c) * HDc + d] = s;
}

// ---------------- scan pass 2: prefix + local scan, divide by k', emit bf16 ----------------
__global__ __launch_bounds__(128) void scan_final_kernel(const float* __restrict__ dotv,
                                                         const float* __restrict__ Vf,
                                                         const float* __restrict__ Kpf,
                                                         const float* __restrict__ psum,
                                                         bf16_t* __restrict__ attn) {
  const int blk = blockIdx.x;
  const int c  = blk % NC;
  const int bh = blk / NC;
  const int b  = bh / Hc;
  const int h  = bh % Hc;
  const int d  = threadIdx.x;
  float state = 0.f;
  for (int cp = 0; cp < c; ++cp)
    state += psum[((size_t)bh * NC + cp) * HDc + d];
  const int tbase = c * CHUNK;
#pragma unroll 2
  for (int tt = 0; tt < CHUNK; ++tt) {
    const int t = tbase + tt;
    const size_t idx = (size_t)(b * Tc + t) * Dc + h * HDc + d;
    state += dotv[(size_t)bh * Tc + t] * Vf[idx];
    attn[idx] = (bf16_t)(state / Kpf[idx]);
  }
}

// ---------------- host orchestration ----------------
extern "C" void kernel_launch(void* const* d_in, const int* in_sizes, int n_in,
                              void* d_out, int out_size, void* d_ws, size_t ws_size,
                              hipStream_t stream) {
  const float* hidden = (const float*)d_in[0];
  const float* Wq = (const float*)d_in[1];
  const float* Wk = (const float*)d_in[2];
  const float* Wv = (const float*)d_in[3];
  const float* Wo = (const float*)d_in[4];
  const float* Fq = (const float*)d_in[5];
  const float* Fk = (const float*)d_in[6];
  float* out = (float*)d_out;

  // workspace carve (all sizes 256B-aligned by construction)
  char* p = (char*)d_ws;
  const size_t tokD_bf = (size_t)NTc * Dc * sizeof(bf16_t);   // 32 MB
  const size_t tokD_f  = (size_t)NTc * Dc * sizeof(float);    // 64 MB
  const size_t wD_bf   = (size_t)Dc * Dc * sizeof(bf16_t);    //  8 MB

  bf16_t* hbf  = (bf16_t*)p; p += tokD_bf;
  bf16_t* WqT  = (bf16_t*)p; p += wD_bf;
  bf16_t* WkT  = (bf16_t*)p; p += wD_bf;
  bf16_t* WvT  = (bf16_t*)p; p += wD_bf;
  bf16_t* WoT  = (bf16_t*)p; p += wD_bf;
  bf16_t* FqT  = (bf16_t*)p; p += wD_bf;
  bf16_t* FkT  = (bf16_t*)p; p += wD_bf;
  bf16_t* Qbf  = (bf16_t*)p; p += tokD_bf;
  bf16_t* Kbf  = (bf16_t*)p; p += tokD_bf;
  float*  Vf   = (float*)p;  p += tokD_f;
  float*  Qpf  = (float*)p;  p += tokD_f;
  float*  Kpf  = (float*)p;  p += tokD_f;
  float*  dotv = (float*)p;  p += (size_t)Bc * Hc * Tc * sizeof(float);
  float*  psum = (float*)p;  p += (size_t)Bc * Hc * NC * HDc * sizeof(float);
  bf16_t* attn = (bf16_t*)p; p += tokD_bf;

  // 1) converts
  {
    int n4 = NTc * Dc / 4;
    cvt_bf16_kernel<<<(n4 + 255) / 256, 256, 0, stream>>>(hidden, hbf, n4);
    dim3 g(Dc / 32, Dc / 32);
    cvtT_bf16_kernel<<<g, 256, 0, stream>>>(Wq, WqT, Dc);
    cvtT_bf16_kernel<<<g, 256, 0, stream>>>(Wk, WkT, Dc);
    cvtT_bf16_kernel<<<g, 256, 0, stream>>>(Wv, WvT, Dc);
    cvtT_bf16_kernel<<<g, 256, 0, stream>>>(Wo, WoT, Dc);
    cvtT_bf16_kernel<<<g, 256, 0, stream>>>(Fq, FqT, Dc);
    cvtT_bf16_kernel<<<g, 256, 0, stream>>>(Fk, FkT, Dc);
  }

  dim3 ggemm(Dc / 128, NTc / 128);   // (16, 64)

  // 2) Q,K (bf16 out), V (f32 out)
  gemm_bf16_wmma<1><<<ggemm, 256, 0, stream>>>(hbf, WqT, Qbf, NTc, Dc, Dc, 0.f);
  gemm_bf16_wmma<1><<<ggemm, 256, 0, stream>>>(hbf, WkT, Kbf, NTc, Dc, Dc, 0.f);
  gemm_bf16_wmma<0><<<ggemm, 256, 0, stream>>>(hbf, WvT, Vf,  NTc, Dc, Dc, 0.f);

  // 3) feature maps fused into GEMM epilogue: exp(x + OFFSET)
  gemm_bf16_wmma<2><<<ggemm, 256, 0, stream>>>(Qbf, FqT, Qpf, NTc, Dc, Dc, OFFSETF);
  gemm_bf16_wmma<2><<<ggemm, 256, 0, stream>>>(Kbf, FkT, Kpf, NTc, Dc, Dc, OFFSETF);

  // 4) per-(b,h,t) dot product
  dot_kernel<<<(Bc * Hc * Tc) / 8, 256, 0, stream>>>(Qpf, Kpf, dotv);

  // 5) blocked causal scan + normalize by k' -> bf16 attn
  scan_partial_kernel<<<Bc * Hc * NC, 128, 0, stream>>>(dotv, Vf, psum);
  scan_final_kernel<<<Bc * Hc * NC, 128, 0, stream>>>(dotv, Vf, Kpf, psum, attn);

  // 6) out = attn @ Wo (f32)
  gemm_bf16_wmma<0><<<ggemm, 256, 0, stream>>>(attn, WoT, out, NTc, Dc, Dc, 0.f);
}
